// KnowMDD_9079560863945
// MI455X (gfx1250) — compile-verified, matrix operand order checked by reading, not run on hardware
//
#include <hip/hip_runtime.h>

// ---------------------------------------------------------------------------
// Types for CDNA5 WMMA
// ---------------------------------------------------------------------------
typedef __attribute__((ext_vector_type(16))) __bf16 bf16x16;
typedef __attribute__((ext_vector_type(8)))  __bf16 bf16x8;
typedef __attribute__((ext_vector_type(4)))  __bf16 bf16x4;
typedef __attribute__((ext_vector_type(8)))  float  f32x8;
typedef __attribute__((ext_vector_type(4)))  float  f32x4;

static __device__ inline __bf16 f2bf(float f) {
  union { float f; unsigned u; } a; a.f = f;
  unsigned r = a.u + 0x7FFFu + ((a.u >> 16) & 1u);   // round-to-nearest-even
  union { unsigned short s; __bf16 b; } o; o.s = (unsigned short)(r >> 16);
  return o.b;
}

static __device__ inline float mishf(float x) {
  float sp = (x > 20.f) ? x : __logf(1.f + __expf(x));
  return x * tanhf(sp);
}

static __device__ inline bf16x16 load_frag(const __bf16* p) {
  bf16x8 lo = *(const bf16x8*)(p);
  bf16x8 hi = *(const bf16x8*)(p + 16);
  return __builtin_shufflevector(lo, hi, 0, 1, 2, 3, 4, 5, 6, 7,
                                 8, 9, 10, 11, 12, 13, 14, 15);
}

static __device__ inline int imin(int a, int b) { return a < b ? a : b; }

// ---------------------------------------------------------------------------
// Generic batched GEMM: C = act(alpha * A@B(^T) + bias), fully specialized at
// compile time over (TRANSB, AASYNC, BASYNC).
// A: [M,K] f32 (lda), B: [K,N] f32 (ldb) or [N,K] if TRANSB, C: [M,N] (ldc)
// 256 thr = 8 wave32; block tile 64x64; each wave computes 16x32 (2 WMMA,
// shared A fragment). Double-buffered LDS pipeline: tiles DMA'd with
// global_load_async_to_lds_b128 (ASYNCcnt), next stage kept in flight via
// s_wait_asynccnt, owner-thread converts fp32->bf16, 1 barrier per K-step.
// ---------------------------------------------------------------------------
template <int TRANSB, int AASYNC, int BASYNC>
__global__ void k_gemm_wmma(const float* __restrict__ A,
                            const float* __restrict__ B,
                            const float* __restrict__ bias,
                            float* __restrict__ C,
                            int M, int N, int K,
                            int lda, int ldb, int ldc,
                            float alpha, int act,
                            long sA, long sB, long sC) {
  A += (long)blockIdx.z * sA;
  B += (long)blockIdx.z * sB;
  C += (long)blockIdx.z * sC;

  const int BM = 64, BN = 64, BK = 32, LD = 40;     // 80B bf16 rows, 16B aligned
  __shared__ __align__(16) __bf16 sAT[2][BM * LD];  // 2 x 5120 B
  __shared__ __align__(16) __bf16 sBT[2][BN * LD];  // 2 x 5120 B
  __shared__ __align__(16) float  sAF[2][BM * BK];  // 2 x 8192 B async staging
  __shared__ __align__(16) float  sBF[2][BN * BK];  // 2 x 8192 B async staging

  const int tid  = threadIdx.x;
  const int lane = tid & 31;
  const int wave = tid >> 5;
  const int wr = wave & 3;
  const int wc = wave >> 2;
  const int rowBase = blockIdx.y * BM;
  const int colBase = blockIdx.x * BN;
  const int laneHalf = lane >> 4;
  const int laneMod  = lane & 15;

  auto issueA = [&](int k0, int s) {
    if constexpr (AASYNC) {
      unsigned ldsA = (unsigned)(size_t)&sAF[s][0];
#pragma unroll
      for (int j = 0; j < 2; j++) {
        int cidx = tid + j * 256;               // 512 chunks: 8 per 32-float row
        int r = cidx >> 3, c4 = (cidx & 7) * 4;
        long gofs = (long)imin(rowBase + r, M - 1) * lda + imin(k0 + c4, lda - 4);
        unsigned long ga = (unsigned long)(const void*)(A + gofs);
        unsigned      la = ldsA + (unsigned)cidx * 16u;
        asm volatile("global_load_async_to_lds_b128 %0, %1, off"
                     :: "v"(la), "v"(ga) : "memory");
      }
    }
  };
  auto issueB = [&](int k0, int s) {
    if constexpr (BASYNC) {
      unsigned ldsB = (unsigned)(size_t)&sBF[s][0];
#pragma unroll
      for (int j = 0; j < 2; j++) {
        int cidx = tid + j * 256;
        long gofs;
        if constexpr (TRANSB) {                 // B[N,K]: 64 rows x 8 chunks
          int n = cidx >> 3, k4 = (cidx & 7) * 4;
          gofs = (long)imin(colBase + n, N - 1) * ldb + imin(k0 + k4, ldb - 4);
        } else {                                // B[K,N]: 32 rows x 16 chunks
          int kk = cidx >> 4, n4 = (cidx & 15) * 4;
          gofs = (long)imin(k0 + kk, K - 1) * ldb + imin(colBase + n4, ldb - 4);
        }
        unsigned long ga = (unsigned long)(const void*)(B + gofs);
        unsigned      la = ldsB + (unsigned)cidx * 16u;
        asm volatile("global_load_async_to_lds_b128 %0, %1, off"
                     :: "v"(la), "v"(ga) : "memory");
      }
    }
  };

  auto convertA = [&](int k0, int s) {
    const bool interior = (rowBase + BM <= M) && (k0 + BK <= K);
    if constexpr (AASYNC) {
#pragma unroll
      for (int j = 0; j < 2; j++) {
        int cidx = tid + j * 256;
        int r = cidx >> 3, c4 = (cidx & 7) * 4;
        f32x4 v = *(const f32x4*)&sAF[s][cidx * 4];
        if (!interior) {
          int gr = rowBase + r;
#pragma unroll
          for (int t = 0; t < 4; t++)
            if (gr >= M || (k0 + c4 + t) >= K) v[t] = 0.f;
        }
        bf16x4 h;
#pragma unroll
        for (int t = 0; t < 4; t++) h[t] = f2bf(v[t]);
        *(bf16x4*)&sAT[s][r * LD + c4] = h;
      }
    } else {
#pragma unroll
      for (int i = 0; i < 8; i++) {
        int idx = tid + i * 256;
        int r = idx >> 5, c = idx & 31;
        int gr = rowBase + r, gk = k0 + c;
        float v;
        if (interior) {
          v = A[(long)gr * lda + gk];
        } else {
          v = A[(long)imin(gr, M - 1) * lda + imin(gk, K - 1)];
          v = (gr < M && gk < K) ? v : 0.f;
        }
        sAT[s][r * LD + c] = f2bf(v);
      }
    }
  };
  auto convertB = [&](int k0, int s) {
    const bool interior = (colBase + BN <= N) && (k0 + BK <= K);
    if constexpr (BASYNC) {
      if constexpr (TRANSB) {
#pragma unroll
        for (int j = 0; j < 2; j++) {
          int cidx = tid + j * 256;
          int n = cidx >> 3, k4 = (cidx & 7) * 4;
          f32x4 v = *(const f32x4*)&sBF[s][cidx * 4];
          if (!interior) {
            int gn = colBase + n;
#pragma unroll
            for (int t = 0; t < 4; t++)
              if (gn >= N || (k0 + k4 + t) >= K) v[t] = 0.f;
          }
          bf16x4 h;
#pragma unroll
          for (int t = 0; t < 4; t++) h[t] = f2bf(v[t]);
          *(bf16x4*)&sBT[s][n * LD + k4] = h;
        }
      } else {
#pragma unroll
        for (int j = 0; j < 2; j++) {
          int cidx = tid + j * 256;
          int kk = cidx >> 4, n4 = (cidx & 15) * 4;
          f32x4 v = *(const f32x4*)&sBF[s][cidx * 4];
          if (!interior) {
            int gk = k0 + kk;
#pragma unroll
            for (int t = 0; t < 4; t++)
              if (gk >= K || (colBase + n4 + t) >= N) v[t] = 0.f;
          }
#pragma unroll
          for (int t = 0; t < 4; t++) sBT[s][(n4 + t) * LD + kk] = f2bf(v[t]);
        }
      }
    } else {
#pragma unroll
      for (int i = 0; i < 8; i++) {
        int idx = tid + i * 256;
        int kk = idx >> 6, n = idx & 63;
        int gk = k0 + kk, gn = colBase + n;
        float v;
        if (interior) {
          v = TRANSB ? B[(long)gn * ldb + gk] : B[(long)gk * ldb + gn];
        } else {
          long bi = TRANSB ? (long)imin(gn, N - 1) * ldb + imin(gk, K - 1)
                           : (long)imin(gk, K - 1) * ldb + imin(gn, N - 1);
          v = B[bi];
          v = (gk < K && gn < N) ? v : 0.f;
        }
        sBT[s][n * LD + kk] = f2bf(v);
      }
    }
  };

  f32x8 acc0 = {0.f, 0.f, 0.f, 0.f, 0.f, 0.f, 0.f, 0.f};
  f32x8 acc1 = {0.f, 0.f, 0.f, 0.f, 0.f, 0.f, 0.f, 0.f};

  issueA(0, 0);
  issueB(0, 0);
  int p = 0;
  for (int k0 = 0; k0 < K; k0 += BK, p ^= 1) {
    const bool hasNext = (k0 + BK) < K;
    if (hasNext) { issueA(k0 + BK, p ^ 1); issueB(k0 + BK, p ^ 1); }
    if constexpr (AASYNC || BASYNC) {
      constexpr int TH = (AASYNC ? 2 : 0) + (BASYNC ? 2 : 0);
      if (hasNext) {
        if constexpr (TH == 4) asm volatile("s_wait_asynccnt 0x4" ::: "memory");
        else                   asm volatile("s_wait_asynccnt 0x2" ::: "memory");
      } else {
        asm volatile("s_wait_asynccnt 0x0" ::: "memory");
      }
    }
    convertA(k0, p);
    convertB(k0, p);
    __syncthreads();

    bf16x16 fa  = load_frag(&sAT[p][(wr * 16 + laneMod) * LD + laneHalf * 8]);
    bf16x16 fb0 = load_frag(&sBT[p][(wc * 32 + laneMod) * LD + laneHalf * 8]);
    bf16x16 fb1 = load_frag(&sBT[p][(wc * 32 + 16 + laneMod) * LD + laneHalf * 8]);
    acc0 = __builtin_amdgcn_wmma_f32_16x16x32_bf16(false, fa, false, fb0,
                                                   (short)0, acc0, false, false);
    acc1 = __builtin_amdgcn_wmma_f32_16x16x32_bf16(false, fa, false, fb1,
                                                   (short)0, acc1, false, false);
  }

  int crow0 = rowBase + wr * 16 + laneHalf * 8;
  int ccol0 = colBase + wc * 32 + laneMod;
  int ccol1 = ccol0 + 16;
  if (ccol0 < N) {
    float bv = bias ? bias[ccol0] : 0.f;
#pragma unroll
    for (int v = 0; v < 8; v++) {
      int r = crow0 + v;
      if (r < M) {
        float o = acc0[v] * alpha + bv;
        if (act == 1) o = mishf(o);
        C[(long)r * ldc + ccol0] = o;
      }
    }
  }
  if (ccol1 < N) {
    float bv = bias ? bias[ccol1] : 0.f;
#pragma unroll
    for (int v = 0; v < 8; v++) {
      int r = crow0 + v;
      if (r < M) {
        float o = acc1[v] * alpha + bv;
        if (act == 1) o = mishf(o);
        C[(long)r * ldc + ccol1] = o;
      }
    }
  }
}

// ---------------------------------------------------------------------------
// Small helper kernels
// ---------------------------------------------------------------------------
__global__ void k_zero(float* p, long n) {
  long i = (long)blockIdx.x * 256 + threadIdx.x;
  if (i < n) p[i] = 0.f;
}

__global__ void k_deg(const int* __restrict__ dst, int nE, float* deg) {
  int e = blockIdx.x * 256 + threadIdx.x;
  if (e < nE) atomicAdd(&deg[dst[e]], 1.f);
}

__global__ void k_dinv(float* deg, int n) {
  int i = blockIdx.x * 256 + threadIdx.x;
  if (i < n) deg[i] = rsqrtf(deg[i] + 1.f);
}

__global__ void k_scatter(const int* __restrict__ src, const int* __restrict__ dst,
                          const float* __restrict__ hw, const float* __restrict__ dinv,
                          float* __restrict__ agg, int nE, int F) {
  int e = blockIdx.x * 4 + (threadIdx.x >> 6);   // 4 edges per 256-thread block
  if (e >= nE) return;
  int f = threadIdx.x & 63;
  int s = src[e], d = dst[e];
  float c = dinv[s] * dinv[d];
  atomicAdd(&agg[(long)d * F + f], hw[(long)s * F + f] * c);
}

__global__ void k_combine(const float* __restrict__ agg, const float* __restrict__ hw,
                          const float* __restrict__ dinv, const float* __restrict__ b,
                          float* __restrict__ out, long n, int F, int act) {
  long idx = (long)blockIdx.x * 256 + threadIdx.x;
  if (idx >= n * F) return;
  long i = idx / F;
  int  f = (int)(idx - i * F);
  float di = dinv[i];
  float o = agg[idx] + hw[idx] * di * di + b[f];
  if (act == 1) o = mishf(o);
  out[idx] = o;
}

__global__ void k_gather(const float* __restrict__ X, const int* __restrict__ idx,
                         float* __restrict__ Y, int F) {
  int r = blockIdx.x;
  int f = threadIdx.x;
  Y[(long)r * F + f] = X[(long)idx[r] * F + f];
}

__global__ void k_pool(const float* __restrict__ X, float* __restrict__ Y,
                       int rowsPer, int F) {
  int g = blockIdx.x, f = threadIdx.x;
  float s = 0.f;
  const float* p = X + (long)g * rowsPer * F + f;
  for (int r = 0; r < rowsPer; r++) s += p[(long)r * F];
  Y[(long)g * F + f] = s / (float)rowsPer;
}

__global__ void k_softmax_rows(float* __restrict__ X, int n) {
  long row = blockIdx.x;
  float* p = X + row * (long)n;
  __shared__ float red[256];
  int tid = threadIdx.x;
  float m = -1e30f;
  for (int i = tid; i < n; i += 256) m = fmaxf(m, p[i]);
  red[tid] = m; __syncthreads();
  for (int s = 128; s > 0; s >>= 1) { if (tid < s) red[tid] = fmaxf(red[tid], red[tid + s]); __syncthreads(); }
  m = red[0]; __syncthreads();
  float sum = 0.f;
  for (int i = tid; i < n; i += 256) { float e = __expf(p[i] - m); p[i] = e; sum += e; }
  red[tid] = sum; __syncthreads();
  for (int s = 128; s > 0; s >>= 1) { if (tid < s) red[tid] += red[tid + s]; __syncthreads(); }
  float inv = 1.f / red[0];
  for (int i = tid; i < n; i += 256) p[i] *= inv;
}

__global__ void k_aw_mean(const float* __restrict__ S, float* __restrict__ out,
                          int G, int RR) {
  int j = blockIdx.x * 256 + threadIdx.x;
  if (j >= RR) return;
  float s = 0.f;
  for (int g = 0; g < G; g++) s += S[(long)g * RR + j];
  out[j] = s / (float)G;
}

__global__ void k_rownorm(const float* __restrict__ X, float* __restrict__ Y, int F) {
  int r = blockIdx.x, f = threadIdx.x;
  __shared__ float red[64];
  float v = X[(long)r * F + f];
  red[f] = v * v; __syncthreads();
  for (int s = 32; s > 0; s >>= 1) { if (f < s) red[f] += red[f + s]; __syncthreads(); }
  float nrm = fmaxf(sqrtf(red[0]), 1e-12f);
  Y[(long)r * F + f] = v / nrm;
}

__global__ void k_closs(const float* __restrict__ L, int G, float* loss) {
  int g = blockIdx.x;
  const float* p = L + (long)g * G;
  __shared__ float red[256];
  int tid = threadIdx.x;
  float m = -1e30f;
  for (int i = tid; i < G; i += 256) m = fmaxf(m, p[i]);
  red[tid] = m; __syncthreads();
  for (int s = 128; s > 0; s >>= 1) { if (tid < s) red[tid] = fmaxf(red[tid], red[tid + s]); __syncthreads(); }
  m = red[0]; __syncthreads();
  float sum = 0.f;
  for (int i = tid; i < G; i += 256) sum += __expf(p[i] - m);
  red[tid] = sum; __syncthreads();
  for (int s = 128; s > 0; s >>= 1) { if (tid < s) red[tid] += red[tid + s]; __syncthreads(); }
  if (tid == 0) {
    float lse = m + __logf(red[0]);
    atomicAdd(loss, -(p[g] - lse) / (float)G);
  }
}

__global__ void k_concat(const float* __restrict__ s1, const float* __restrict__ z1,
                         const float* __restrict__ s2, const float* __restrict__ z2,
                         float* __restrict__ feat) {
  int g = blockIdx.x, t = threadIdx.x;
  const float* src = (t < 64) ? s1 : (t < 128) ? z1 : (t < 192) ? s2 : z2;
  int f = t & 63;
  feat[(long)g * 256 + t] = src[(long)g * 64 + f];
}

// ---------------------------------------------------------------------------
// Orchestration
// ---------------------------------------------------------------------------
extern "C" void kernel_launch(void* const* d_in, const int* in_sizes, int n_in,
                              void* d_out, int out_size, void* d_ws, size_t ws_size,
                              hipStream_t stream) {
  (void)in_sizes; (void)n_in; (void)out_size; (void)ws_size;

  const int G = 512, R1 = 200, R2 = 160, FIN = 128, H = 64, OUT = 64, DEG = 16, V = 64;
  const int N1 = G * R1, N2 = G * R2;
  const int E1 = N1 * DEG, E2 = N2 * DEG;

  const float* x1      = (const float*)d_in[0];
  const float* x2      = (const float*)d_in[1];
  const float* fcl_W   = (const float*)d_in[2];  const float* fcl_b   = (const float*)d_in[3];
  const float* fcr_W   = (const float*)d_in[4];  const float* fcr_b   = (const float*)d_in[5];
  const float* fcl1_W  = (const float*)d_in[6];  const float* fcl1_b  = (const float*)d_in[7];
  const float* fcr1_W  = (const float*)d_in[8];  const float* fcr1_b  = (const float*)d_in[9];
  const float* a1q_W   = (const float*)d_in[10]; const float* a1q_b   = (const float*)d_in[11];
  const float* a1k_W   = (const float*)d_in[12]; const float* a1k_b   = (const float*)d_in[13];
  const float* a1v_W   = (const float*)d_in[14]; const float* a1v_b   = (const float*)d_in[15];
  const float* a2q_W   = (const float*)d_in[16]; const float* a2q_b   = (const float*)d_in[17];
  const float* a2k_W   = (const float*)d_in[18]; const float* a2k_b   = (const float*)d_in[19];
  const float* a2v_W   = (const float*)d_in[20]; const float* a2v_b   = (const float*)d_in[21];
  const float* conv_W  = (const float*)d_in[22]; const float* conv_b  = (const float*)d_in[23];
  const float* conv1_W = (const float*)d_in[24]; const float* conv1_b = (const float*)d_in[25];
  const float* mlp_W   = (const float*)d_in[26]; const float* mlp_b   = (const float*)d_in[27];
  const float* fc1a_W  = (const float*)d_in[28]; const float* fc1a_b  = (const float*)d_in[29];
  const float* fc1b_W  = (const float*)d_in[30]; const float* fc1b_b  = (const float*)d_in[31];
  const int* ei1  = (const int*)d_in[32];
  const int* ei2  = (const int*)d_in[33];
  const int* sub1 = (const int*)d_in[36];
  const int* sub2 = (const int*)d_in[37];
  const int* src1 = ei1;       const int* dst1 = ei1 + E1;
  const int* src2 = ei2;       const int* dst2 = ei2 + E2;

  float* out = (float*)d_out;
  float* aw1_out = out + 1 + G * 2;
  float* aw2_out = aw1_out + R1 * R1;

  float* W = (float*)d_ws;
  size_t off = 0;
  auto alloc = [&](size_t n) { float* p = W + off; off += (n + 63) & ~(size_t)63; return p; };

  float* z1pool = alloc((size_t)G * OUT);
  float* z2pool = alloc((size_t)G * OUT);
  float* s1pool = alloc((size_t)G * OUT);
  float* s2pool = alloc((size_t)G * OUT);
  float* an     = alloc((size_t)G * OUT);
  float* bn     = alloc((size_t)G * OUT);
  float* feat   = alloc((size_t)G * 256);
  float* f1     = alloc((size_t)G * FIN);
  const size_t persistentEnd = off;

  auto gemm = [&](const float* A, const float* B, const float* bias, float* C,
                  int M, int N, int K, int lda, int ldb, int ldc,
                  int transB, float alpha, int act,
                  int batch = 1, long sA = 0, long sB = 0, long sC = 0) {
    dim3 g((N + 63) / 64, (M + 63) / 64, batch);
    const bool aA = ((lda & 3) == 0);
    const bool bA = ((ldb & 3) == 0);
#define GEMM_CALL(T, AA, BB)                                                   \
    k_gemm_wmma<T, AA, BB><<<g, 256, 0, stream>>>(                             \
        A, B, bias, C, M, N, K, lda, ldb, ldc, alpha, act, sA, sB, sC)
    if (transB) {
      if (aA && bA)      GEMM_CALL(1, 1, 1);
      else if (aA)       GEMM_CALL(1, 1, 0);
      else if (bA)       GEMM_CALL(1, 0, 1);
      else               GEMM_CALL(1, 0, 0);
    } else {
      if (aA && bA)      GEMM_CALL(0, 1, 1);
      else if (aA)       GEMM_CALL(0, 1, 0);
      else if (bA)       GEMM_CALL(0, 0, 1);
      else               GEMM_CALL(0, 0, 0);
    }
#undef GEMM_CALL
  };
  auto zero = [&](float* p, long n) {
    k_zero<<<(unsigned)((n + 255) / 256), 256, 0, stream>>>(p, n);
  };

  const float invsq = 0.125f;   // 1/sqrt(64)
  const float invT  = 1.f / 0.6f;

  // ======================= attention, modality 1 =======================
  {
    off = persistentEnd;
    float* zpre = alloc((size_t)N1 * FIN);
    float* Q    = alloc((size_t)N1 * OUT);
    float* Km   = alloc((size_t)N1 * OUT);
    float* Vm   = alloc((size_t)N1 * OUT);
    float* S    = alloc((size_t)G * R1 * R1);
    float* zatt = alloc((size_t)N1 * OUT);

    gemm(x1, fcl1_W, fcl1_b, zpre, N1, FIN, R1, R1 + 3, FIN, FIN, 0, 1.f, 0);
    gemm(zpre, a1q_W, a1q_b, Q,  N1, OUT, FIN, FIN, OUT, OUT, 0, 1.f, 0);
    gemm(zpre, a1k_W, a1k_b, Km, N1, OUT, FIN, FIN, OUT, OUT, 0, 1.f, 0);
    gemm(zpre, a1v_W, a1v_b, Vm, N1, OUT, FIN, FIN, OUT, OUT, 0, 1.f, 0);
    gemm(Q, Km, nullptr, S, R1, R1, OUT, OUT, OUT, R1, 1, invsq, 0,
         G, (long)R1 * OUT, (long)R1 * OUT, (long)R1 * R1);
    k_softmax_rows<<<G * R1, 256, 0, stream>>>(S, R1);
    k_aw_mean<<<(R1 * R1 + 255) / 256, 256, 0, stream>>>(S, aw1_out, G, R1 * R1);
    gemm(S, Vm, nullptr, zatt, R1, OUT, R1, R1, OUT, OUT, 0, 1.f, 0,
         G, (long)R1 * R1, (long)R1 * OUT, (long)R1 * OUT);
    k_pool<<<G, OUT, 0, stream>>>(zatt, z1pool, R1, OUT);
  }

  // ======================= attention, modality 2 =======================
  {
    off = persistentEnd;
    float* zpre = alloc((size_t)N2 * FIN);
    float* Q    = alloc((size_t)N2 * OUT);
    float* Km   = alloc((size_t)N2 * OUT);
    float* Vm   = alloc((size_t)N2 * OUT);
    float* S    = alloc((size_t)G * R2 * R2);
    float* zatt = alloc((size_t)N2 * OUT);

    gemm(x2, fcr1_W, fcr1_b, zpre, N2, FIN, R2, R2 + 3, FIN, FIN, 0, 1.f, 0);
    gemm(zpre, a2q_W, a2q_b, Q,  N2, OUT, FIN, FIN, OUT, OUT, 0, 1.f, 0);
    gemm(zpre, a2k_W, a2k_b, Km, N2, OUT, FIN, FIN, OUT, OUT, 0, 1.f, 0);
    gemm(zpre, a2v_W, a2v_b, Vm, N2, OUT, FIN, FIN, OUT, OUT, 0, 1.f, 0);
    gemm(Q, Km, nullptr, S, R2, R2, OUT, OUT, OUT, R2, 1, invsq, 0,
         G, (long)R2 * OUT, (long)R2 * OUT, (long)R2 * R2);
    k_softmax_rows<<<G * R2, 256, 0, stream>>>(S, R2);
    k_aw_mean<<<(R2 * R2 + 255) / 256, 256, 0, stream>>>(S, aw2_out, G, R2 * R2);
    gemm(S, Vm, nullptr, zatt, R2, OUT, R2, R2, OUT, OUT, 0, 1.f, 0,
         G, (long)R2 * R2, (long)R2 * OUT, (long)R2 * OUT);
    k_pool<<<G, OUT, 0, stream>>>(zatt, z2pool, R2, OUT);
  }

  // ======================= GCN path (per modality) =======================
  auto gcn_path = [&](const float* x, int N, int inF, const float* fcW, const float* fcb,
                      const int* src, const int* dst, int nE, const int* sub,
                      float* spool) {
    off = persistentEnd;
    float* h    = alloc((size_t)N * FIN);
    float* hw   = alloc((size_t)N * H);
    float* dinv = alloc((size_t)N);
    float* agg  = alloc((size_t)N * H);
    float* o1   = alloc((size_t)N * H);
    float* hw2  = alloc((size_t)N * H);
    float* agg2 = alloc((size_t)N * H);
    float* o2   = alloc((size_t)N * H);
    float* gat  = alloc((size_t)G * V * H);
    float* srow = alloc((size_t)G * V * OUT);

    gemm(x, fcW, fcb, h, N, FIN, inF, inF, FIN, FIN, 0, 1.f, 0);

    zero(dinv, N);
    k_deg<<<(nE + 255) / 256, 256, 0, stream>>>(dst, nE, dinv);
    k_dinv<<<(N + 255) / 256, 256, 0, stream>>>(dinv, N);

    gemm(h, conv_W, nullptr, hw, N, H, FIN, FIN, H, H, 0, 1.f, 0);
    zero(agg, (long)N * H);
    k_scatter<<<(nE + 3) / 4, 256, 0, stream>>>(src, dst, hw, dinv, agg, nE, H);
    k_combine<<<(unsigned)(((long)N * H + 255) / 256), 256, 0, stream>>>(
        agg, hw, dinv, conv_b, o1, N, H, 1);

    gemm(o1, conv1_W, nullptr, hw2, N, H, H, H, H, H, 0, 1.f, 0);
    zero(agg2, (long)N * H);
    k_scatter<<<(nE + 3) / 4, 256, 0, stream>>>(src, dst, hw2, dinv, agg2, nE, H);
    k_combine<<<(unsigned)(((long)N * H + 255) / 256), 256, 0, stream>>>(
        agg2, hw2, dinv, conv1_b, o2, N, H, 0);

    k_gather<<<G * V, H, 0, stream>>>(o2, sub, gat, H);
    gemm(gat, mlp_W, mlp_b, srow, G * V, OUT, H, H, OUT, OUT, 0, 1.f, 0);
    k_pool<<<G, OUT, 0, stream>>>(srow, spool, V, OUT);
  };

  gcn_path(x1, N1, R1 + 3, fcl_W, fcl_b, src1, dst1, E1, sub1, s1pool);
  gcn_path(x2, N2, R2 + 3, fcr_W, fcr_b, src2, dst2, E2, sub2, s2pool);

  // ======================= contrastive loss =======================
  zero(out, 1);
  {
    off = persistentEnd;
    float* logits = alloc((size_t)G * G);

    k_rownorm<<<G, 64, 0, stream>>>(s1pool, an, OUT);
    k_rownorm<<<G, 64, 0, stream>>>(z2pool, bn, OUT);
    gemm(an, bn, nullptr, logits, G, G, OUT, OUT, OUT, G, 1, invT, 0);
    k_closs<<<G, 256, 0, stream>>>(logits, G, out);

    k_rownorm<<<G, 64, 0, stream>>>(z1pool, an, OUT);
    k_rownorm<<<G, 64, 0, stream>>>(s2pool, bn, OUT);
    gemm(an, bn, nullptr, logits, G, G, OUT, OUT, OUT, G, 1, invT, 0);
    k_closs<<<G, 256, 0, stream>>>(logits, G, out);
  }

  // ======================= classifier head =======================
  k_concat<<<G, 256, 0, stream>>>(s1pool, z1pool, s2pool, z2pool, feat);
  gemm(feat, fc1a_W, fc1a_b, f1, G, FIN, 4 * OUT, 4 * OUT, FIN, FIN, 0, 1.f, 1);
  gemm(f1, fc1b_W, fc1b_b, out + 1, G, 2, FIN, FIN, 2, 2, 0, 1.f, 0);
}
